// RealMultiheadAttention_90228672955117
// MI455X (gfx1250) — matrix operand order repaired and checked by reading, not compiled, not run on hardware
//
#include <hip/hip_runtime.h>
#include <stdint.h>

typedef __attribute__((ext_vector_type(16))) _Float16 v16h;
typedef __attribute__((ext_vector_type(8)))  _Float16 v8h;
typedef __attribute__((ext_vector_type(4)))  _Float16 v4h;
typedef __attribute__((ext_vector_type(8)))  float    v8f;
typedef __attribute__((ext_vector_type(4)))  int      v4i;

static constexpr int Sn  = 2048;
static constexpr int Dn  = 1024;
static constexpr int Hn  = 16;
static constexpr int HDn = 64;
static constexpr int Bn  = 2;

#define WMMA_F16(a, b, c) \
  __builtin_amdgcn_wmma_f32_16x16x32_f16(false, (a), false, (b), (short)0, (c), false, false)

#if defined(__has_builtin)
#if __has_builtin(__builtin_amdgcn_global_load_async_to_lds_b128)
#define MHA_ASYNC 1
#endif
#endif

// 16-byte global -> LDS copy. Async (ASYNCcnt-tracked) when the gfx1250
// builtin exists; otherwise a plain load+ds_store pair.
__device__ __forceinline__ void cp16(void* lds, const void* g) {
#ifdef MHA_ASYNC
  // ISA 10.2: generic LDS address truncates to the LDS offset in [31:0];
  // generic global address is numerically the AS(1) address.
  __builtin_amdgcn_global_load_async_to_lds_b128(
      (__attribute__((address_space(1))) v4i*)(uintptr_t)g,
      (__attribute__((address_space(3))) v4i*)(uint32_t)(uintptr_t)lds,
      0, 0);
#else
  *(v8h*)lds = *(const v8h*)g;
#endif
}

__device__ __forceinline__ void cp16_wait() {
#ifdef MHA_ASYNC
#if __has_builtin(__builtin_amdgcn_s_wait_asynccnt)
  __builtin_amdgcn_s_wait_asynccnt(0);
#else
  asm volatile("s_wait_asynccnt 0x0" ::: "memory");
#endif
#endif
}

__device__ __forceinline__ v16h cat8(v8h lo, v8h hi) {
  v16h r;
#pragma unroll
  for (int i = 0; i < 8; ++i) { r[i] = lo[i]; r[i + 8] = hi[i]; }
  return r;
}

// ---------------------------------------------------------------------------
// fp32 -> f16 pack (one float4 per thread)
// ---------------------------------------------------------------------------
__global__ __launch_bounds__(256)
void mha_pack_f16(const float* __restrict__ src, _Float16* __restrict__ dst,
                  int n4) {
  const int i = blockIdx.x * 256 + threadIdx.x;
  if (i < n4) {
    const float4 f = ((const float4*)src)[i];
    v4h h;
    h[0] = (_Float16)f.x; h[1] = (_Float16)f.y;
    h[2] = (_Float16)f.z; h[3] = (_Float16)f.w;
    ((v4h*)dst)[i] = h;
  }
}

// ---------------------------------------------------------------------------
// GEMM: C[M,N] = A[M,K](f16) * W[N,K](f16)^T + bias[N](f32)
// 256 threads = 8 waves, tile 128x64, K-panel 64, double-buffered async LDS.
// mode 0: f16 row-major out; mode 1: f16 V^T out [b,h,d,s]; mode 2: f32 out.
// ---------------------------------------------------------------------------
__global__ __launch_bounds__(256)
void mha_gemm_f16(const _Float16* __restrict__ A, const _Float16* __restrict__ W,
                  const float* __restrict__ bias, void* __restrict__ outp,
                  int M, int N, int K, int mode) {
  __shared__ _Float16 sA[2][128][64];
  __shared__ _Float16 sB[2][64][64];

  const int tid    = threadIdx.x;
  const int lane   = tid & 31;
  const int wid    = tid >> 5;
  const int l16    = lane & 15;
  const int half16 = lane >> 4;
  const int n0     = blockIdx.x * 64;
  const int m0     = blockIdx.y * 128;

  auto stage = [&](int buf, int kb) {
#pragma unroll
    for (int it = 0; it < 4; ++it) {            // A: 128x64 halves = 1024 x16B
      int slot = it * 256 + tid;
      int row = slot >> 3, c = (slot & 7) << 3;
      cp16(&sA[buf][row][c], &A[(size_t)(m0 + row) * K + kb + c]);
    }
#pragma unroll
    for (int it = 0; it < 2; ++it) {            // W: 64x64 halves = 512 x16B
      int slot = it * 256 + tid;
      int row = slot >> 3, c = (slot & 7) << 3;
      cp16(&sB[buf][row][c], &W[(size_t)(n0 + row) * K + kb + c]);
    }
  };

  v8f acc[4] = {};
  const int nstage = K >> 6;

  stage(0, 0);
  cp16_wait();
  __syncthreads();

  for (int s = 0; s < nstage; ++s) {
    const int buf = s & 1;
    if (s + 1 < nstage) stage(buf ^ 1, (s + 1) << 6);  // overlap with compute

#pragma unroll
    for (int kk = 0; kk < 64; kk += 32) {
      const int abase = kk + half16 * 16;
      v16h af = *(const v16h*)&sA[buf][wid * 16 + l16][abase];
#pragma unroll
      for (int nt = 0; nt < 4; ++nt) {
        v16h bf = *(const v16h*)&sB[buf][nt * 16 + l16][kk + half16 * 16];
        acc[nt] = WMMA_F16(af, bf, acc[nt]);
      }
    }
    cp16_wait();
    __syncthreads();
  }

  // epilogue: C layout -> row = m0+wid*16+r+8*half16, col = n0+nt*16+l16
  if (mode == 0) {
    _Float16* out = (_Float16*)outp;
#pragma unroll
    for (int nt = 0; nt < 4; ++nt) {
      const int n = n0 + nt * 16 + l16;
      const float bv = bias[n];
#pragma unroll
      for (int r = 0; r < 8; ++r) {
        const int row = m0 + wid * 16 + r + half16 * 8;
        out[(size_t)row * N + n] = (_Float16)(acc[nt][r] + bv);
      }
    }
  } else if (mode == 1) {
    _Float16* out = (_Float16*)outp;  // VT[((b*H + h)*64 + d)*S + s]
#pragma unroll
    for (int nt = 0; nt < 4; ++nt) {
      const int n = n0 + nt * 16 + l16;  // = h*64 + d
      const float bv = bias[n];
#pragma unroll
      for (int r = 0; r < 8; ++r) {
        const int row = m0 + wid * 16 + r + half16 * 8;
        const int b = row >> 11, sp = row & (Sn - 1);
        out[(size_t)(b * (Hn * HDn) + n) * Sn + sp] = (_Float16)(acc[nt][r] + bv);
      }
    }
  } else {
    float* out = (float*)outp;
#pragma unroll
    for (int nt = 0; nt < 4; ++nt) {
      const int n = n0 + nt * 16 + l16;
      const float bv = bias[n];
#pragma unroll
      for (int r = 0; r < 8; ++r) {
        const int row = m0 + wid * 16 + r + half16 * 8;
        out[(size_t)row * N + n] = acc[nt][r] + bv;
      }
    }
  }
}

// ---------------------------------------------------------------------------
// Flash attention, all-f16 inputs. One (b,h) x 64-q-row tile per block
// (4 waves). Double-buffered async K / V^T staging; online softmax.
// ---------------------------------------------------------------------------
__global__ __launch_bounds__(128)
void mha_flash_attn(const _Float16* __restrict__ Qh, const _Float16* __restrict__ Kh,
                    const _Float16* __restrict__ VT,
                    const unsigned char* __restrict__ amask,
                    _Float16* __restrict__ Ah) {
  __shared__ _Float16 sK[2][32][64];   // [kpos][d]
  __shared__ _Float16 sVT[2][64][32];  // [d][kpos]
  __shared__ _Float16 sP[4][16][32];   // per-wave P staging

  const int tid    = threadIdx.x;
  const int lane   = tid & 31;
  const int wid    = tid >> 5;
  const int l16    = lane & 15;
  const int half16 = lane >> 4;

  const int q0 = blockIdx.x * 64;
  const int bh = blockIdx.y;
  const int b  = bh / Hn;
  const int h  = bh % Hn;
  const int qBase = q0 + wid * 16;

  auto stage = [&](int buf, int kb) {
#pragma unroll
    for (int it = 0; it < 2; ++it) {            // K tile: 32x64 = 256 x16B
      int slot = it * 128 + tid;
      int kr = slot >> 3, c = (slot & 7) << 3;
      cp16(&sK[buf][kr][c],
           &Kh[(size_t)(b * Sn + kb + kr) * Dn + h * HDn + c]);
    }
#pragma unroll
    for (int it = 0; it < 2; ++it) {            // VT tile: 64x32 = 256 x16B
      int slot = it * 128 + tid;
      int dr = slot >> 2, c = (slot & 3) << 3;
      cp16(&sVT[buf][dr][c], &VT[(size_t)(bh * HDn + dr) * Sn + kb + c]);
    }
  };

  // Q fragments (f16, direct from global), held for the whole loop
  const _Float16* qrow = Qh + (size_t)(b * Sn + qBase + l16) * Dn + h * HDn;
  v16h qf[2];
#pragma unroll
  for (int j = 0; j < 2; ++j) {
    const int base = j * 32 + half16 * 16;
    qf[j] = *(const v16h*)&qrow[base];
  }

  float m_run[8], l_run[8];
  v8f o[4] = {};
#pragma unroll
  for (int r = 0; r < 8; ++r) { m_run[r] = -__builtin_inff(); l_run[r] = 0.f; }

  const float sc   = 0.125f * 1.44269504088896340736f;  // 1/sqrt(64)*log2(e)
  const int  nblk  = (q0 >> 5) + 2;                     // keys 0 .. q0+63

  stage(0, 0);
  cp16_wait();
  __syncthreads();

  for (int blk = 0; blk < nblk; ++blk) {
    const int kb  = blk << 5;
    const int buf = blk & 1;
    if (blk + 1 < nblk) stage(buf ^ 1, (blk + 1) << 5);

    // scores: two 16-key tiles, 2 WMMAs each over d
    v8f s[2];
#pragma unroll
    for (int nt = 0; nt < 2; ++nt) {
      v8f c = {};
#pragma unroll
      for (int j = 0; j < 2; ++j) {
        v16h bf = *(const v16h*)&sK[buf][nt * 16 + l16][j * 32 + half16 * 16];
        c = WMMA_F16(qf[j], bf, c);
      }
      s[nt] = c;
    }

    int   kcol[2];
    float colpen[2];
#pragma unroll
    for (int nt = 0; nt < 2; ++nt) {
      kcol[nt]   = kb + nt * 16 + l16;
      colpen[nt] = (amask[b * Sn + kcol[nt]] != 0) ? 0.f : -__builtin_inff();
    }

#pragma unroll
    for (int r = 0; r < 8; ++r) {
      const int qrowi = qBase + r + half16 * 8;
#pragma unroll
      for (int nt = 0; nt < 2; ++nt) {
        float v = s[nt][r] * sc + colpen[nt];
        if (kcol[nt] > qrowi) v = -__builtin_inff();
        s[nt][r] = v;
      }
      float rm = fmaxf(s[0][r], s[1][r]);
#pragma unroll
      for (int off = 1; off < 16; off <<= 1)
        rm = fmaxf(rm, __shfl_xor(rm, off, 32));
      const float m_new = fmaxf(m_run[r], rm);
      float alpha = exp2f(m_run[r] - m_new);
      if (m_new == -__builtin_inff()) alpha = 0.f;
      float psum = 0.f;
#pragma unroll
      for (int nt = 0; nt < 2; ++nt) {
        float p = (s[nt][r] == -__builtin_inff()) ? 0.f : exp2f(s[nt][r] - m_new);
        s[nt][r] = p;
        psum += p;
      }
#pragma unroll
      for (int off = 1; off < 16; off <<= 1)
        psum += __shfl_xor(psum, off, 32);
      l_run[r] = l_run[r] * alpha + psum;
      m_run[r] = m_new;
#pragma unroll
      for (int dt = 0; dt < 4; ++dt) o[dt][r] *= alpha;
    }

    // P (C layout) -> A fragment via per-wave LDS
#pragma unroll
    for (int nt = 0; nt < 2; ++nt)
#pragma unroll
      for (int r = 0; r < 8; ++r)
        sP[wid][r + half16 * 8][nt * 16 + l16] = (_Float16)s[nt][r];

    {
      v16h pf = *(const v16h*)&sP[wid][l16][half16 * 16];
#pragma unroll
      for (int dt = 0; dt < 4; ++dt) {
        v16h vf = *(const v16h*)&sVT[buf][dt * 16 + l16][half16 * 16];
        o[dt] = WMMA_F16(pf, vf, o[dt]);
      }
    }
    cp16_wait();
    __syncthreads();
  }

  // normalize, store f16 attention output [b, s, h*64 + d]
#pragma unroll
  for (int r = 0; r < 8; ++r) {
    const float inv = (l_run[r] > 0.f) ? 1.f / l_run[r] : 0.f;
    const int qrowi = qBase + r + half16 * 8;
#pragma unroll
    for (int dt = 0; dt < 4; ++dt)
      Ah[(size_t)(b * Sn + qrowi) * Dn + h * HDn + dt * 16 + l16] =
          (_Float16)(o[dt][r] * inv);
  }
}

// ---------------------------------------------------------------------------
extern "C" void kernel_launch(void* const* d_in, const int* in_sizes, int n_in,
                              void* d_out, int out_size, void* d_ws, size_t ws_size,
                              hipStream_t stream) {
  const float*         hidden = (const float*)d_in[0];
  const unsigned char* amask  = (const unsigned char*)d_in[1];
  const float* Wq = (const float*)d_in[2];
  const float* bq = (const float*)d_in[3];
  const float* Wk = (const float*)d_in[4];
  const float* bk = (const float*)d_in[5];
  const float* Wv = (const float*)d_in[6];
  const float* bv = (const float*)d_in[7];
  const float* Wo = (const float*)d_in[8];
  const float* bo = (const float*)d_in[9];
  float* out = (float*)d_out;

  const int M = Bn * Sn;  // 4096
  const size_t actBytes = (size_t)M * Dn * sizeof(_Float16);   // 8 MB
  const size_t wBytes   = (size_t)Dn * Dn * sizeof(_Float16);  // 2 MB
  char* ws = (char*)d_ws;
  _Float16* Xh  = (_Float16*)(ws);
  _Float16* Wqh = (_Float16*)(ws + actBytes);
  _Float16* Wkh = (_Float16*)(ws + actBytes + 1 * wBytes);
  _Float16* Wvh = (_Float16*)(ws + actBytes + 2 * wBytes);
  _Float16* Woh = (_Float16*)(ws + actBytes + 3 * wBytes);
  _Float16* Qh  = (_Float16*)(ws + actBytes + 4 * wBytes);
  _Float16* Kh  = (_Float16*)(ws + 2 * actBytes + 4 * wBytes);
  _Float16* VTh = (_Float16*)(ws + 3 * actBytes + 4 * wBytes);
  _Float16* Ah  = (_Float16*)(ws + 4 * actBytes + 4 * wBytes);

  // fp32 -> f16 packs
  {
    const int n4h = (M * Dn) / 4;
    mha_pack_f16<<<dim3(n4h / 256), dim3(256), 0, stream>>>(hidden, Xh, n4h);
    const int n4w = (Dn * Dn) / 4;
    mha_pack_f16<<<dim3(n4w / 256), dim3(256), 0, stream>>>(Wq, Wqh, n4w);
    mha_pack_f16<<<dim3(n4w / 256), dim3(256), 0, stream>>>(Wk, Wkh, n4w);
    mha_pack_f16<<<dim3(n4w / 256), dim3(256), 0, stream>>>(Wv, Wvh, n4w);
    mha_pack_f16<<<dim3(n4w / 256), dim3(256), 0, stream>>>(Wo, Woh, n4w);
  }

  dim3 gblk(256), ggrd(Dn / 64, M / 128);  // (16, 32)
  mha_gemm_f16<<<ggrd, gblk, 0, stream>>>(Xh, Wqh, bq, Qh,  M, Dn, Dn, 0);
  mha_gemm_f16<<<ggrd, gblk, 0, stream>>>(Xh, Wkh, bk, Kh,  M, Dn, Dn, 0);
  mha_gemm_f16<<<ggrd, gblk, 0, stream>>>(Xh, Wvh, bv, VTh, M, Dn, Dn, 1);

  dim3 ablk(128), agrd(Sn / 64, Bn * Hn);  // (32, 32)
  mha_flash_attn<<<agrd, ablk, 0, stream>>>(Qh, Kh, VTh, amask, Ah);

  mha_gemm_f16<<<ggrd, gblk, 0, stream>>>(Ah, Woh, bo, out, M, Dn, Dn, 2);
}
